// alpha_LSTMLayer_60189671686946
// MI455X (gfx1250) — compile-verified, matrix operand order checked by reading, not run on hardware
//
#include <hip/hip_runtime.h>

// alpha-LSTM layer, persistent-kernel design for MI455X (gfx1250, wave32).
// T=1024 steps, B=64, I=H=512.
//  - one-time: convert x [T,B,I] fp32 -> bf16 in workspace (streaming, ~9us at 23.3TB/s)
//  - persistent kernel, 32 WGs; each WG owns 16 hidden units; its 64 gate rows
//    (i/f/g/o strips) x K=1024 of [W_ih||W_hh] pinned in LDS as bf16 (CDNA5: 320KB/WGP)
//  - per step: gates = [x_t||h](64x1024) @ Wslice^T via v_wmma_f32_16x16x32_bf16,
//    fused sigmoid/tanh/alpha-interp elementwise with c/h state + biases held in
//    REGISTERS (fixed per-thread element mapping), h carried bf16 double-buffered
//    in global memory for the cross-WG GEMM operand
//  - ys written with non-temporal stores (134MB write-once vs 192MB L2; keep x/h resident)
//  - per-step device-wide sync: monotonic atomic-counter barrier (32 WGs co-resident)

#define T_STEPS 1024
#define BATCH   64
#define IDIM    512
#define HDIM    512
#define NWG     32
#define HS      (HDIM / NWG)      // 16 hidden cols per WG
#define KTOT    (IDIM + HDIM)     // 1024
#define WROWS   (4 * HS)          // 64 gate rows per WG
#define WSTRIDE (KTOT + 8)        // padded LDS row stride -> conflict-free ds_load_b128
#define GST     84                // gate-buffer row stride (floats): conflict-free R/W

typedef __bf16 bf16_t;
typedef __attribute__((ext_vector_type(16))) __bf16 v16bf;
typedef __attribute__((ext_vector_type(8)))  __bf16 v8bf;
typedef __attribute__((ext_vector_type(8)))  float  v8f;

static __device__ __forceinline__ unsigned short f2bf_rne(float f) {
  union { float f; unsigned u; } v; v.f = f;
  unsigned r = v.u + 0x7FFFu + ((v.u >> 16) & 1u);
  return (unsigned short)(r >> 16);
}
static __device__ __forceinline__ float fast_sigmoid(float x) {
  return 1.f / (1.f + __expf(-x));            // v_exp_f32 path
}
static __device__ __forceinline__ float fast_tanh(float x) {
  return 2.f / (1.f + __expf(-2.f * x)) - 1.f; // tanh = 2*sigmoid(2x)-1
}

// ---- x fp32 -> bf16 conversion (one pass, 4 elems/thread) ----
__global__ void xcvt_kernel(const float* __restrict__ x,
                            unsigned short* __restrict__ xb, int n4) {
  int i = blockIdx.x * blockDim.x + threadIdx.x;
  if (i < n4) {
    float4 v = reinterpret_cast<const float4*>(x)[i];
    ushort4 o;
    o.x = f2bf_rne(v.x); o.y = f2bf_rne(v.y);
    o.z = f2bf_rne(v.z); o.w = f2bf_rne(v.w);
    reinterpret_cast<ushort4*>(xb)[i] = o;
  }
}

// ---- h0 fp32 -> bf16 into hbuf[0], zero the barrier counter ----
__global__ void hprep_kernel(const float* __restrict__ h0,
                             unsigned short* __restrict__ hbuf0,
                             unsigned int* __restrict__ syncp) {
  int i = blockIdx.x * blockDim.x + threadIdx.x;
  if (i == 0) syncp[0] = 0u;
  if (i < (BATCH * HDIM) / 4) {
    float4 v = reinterpret_cast<const float4*>(h0)[i];
    ushort4 o;
    o.x = f2bf_rne(v.x); o.y = f2bf_rne(v.y);
    o.z = f2bf_rne(v.z); o.w = f2bf_rne(v.w);
    reinterpret_cast<ushort4*>(hbuf0)[i] = o;
  }
}

__launch_bounds__(256, 1)
__global__ void lstm_persistent(const unsigned short* __restrict__ xb,   // [T,B,I] bf16
                                const float* __restrict__ h0,
                                const float* __restrict__ c0,
                                const float* __restrict__ Wih,           // [4H, I]
                                const float* __restrict__ Whh,           // [4H, H]
                                const float* __restrict__ bih,
                                const float* __restrict__ bhh,
                                const float* __restrict__ alphap,
                                unsigned short* __restrict__ hbuf,       // [2][B*H] bf16
                                unsigned int* __restrict__ syncp,
                                float* __restrict__ out)                 // ys | hT | cT
{
  // 64 x (1024+8) bf16 weights = 132,096 B ; 64 x 84 fp32 gate staging = 21,504 B
  __shared__ __attribute__((aligned(16))) unsigned short Wsl[WROWS * WSTRIDE];
  __shared__ __attribute__((aligned(16))) float gbuf[BATCH * GST];

  const int wg    = blockIdx.x;        // 0..31, owns hidden cols [wg*HS, wg*HS+HS)
  const int tid   = threadIdx.x;       // 256 threads = 8 waves
  const int lane  = tid & 31;
  const int wave  = tid >> 5;
  const int mtile = wave & 3;          // batch-row tile (16 rows)
  const int nb    = (wave >> 2) * 2;   // this wave's 2 gate-col tiles: nb, nb+1
  const int lr    = lane & 15;
  const int hi    = lane >> 4;

  // ---- prologue: pin weight slice in LDS (fp32 -> bf16) ----
  for (int e = tid; e < WROWS * KTOT; e += 256) {
    int r = e >> 10, k = e & (KTOT - 1);
    int strip = r >> 4, rr = r & (HS - 1);
    int grow = strip * HDIM + wg * HS + rr;       // global gate row
    float w = (k < IDIM) ? Wih[grow * IDIM + k] : Whh[grow * HDIM + (k - IDIM)];
    Wsl[r * WSTRIDE + k] = f2bf_rne(w);
  }

  // ---- register-resident per-thread state (fixed element mapping) ----
  // thread owns elements (b = b0+16q, j) for q=0..3, with j = tid&15, b0 = tid>>4
  const int j  = tid & 15;
  const int b0 = tid >> 4;                        // 0..15
  const int gc = wg * HS + j;                     // global hidden col
  const float bi_i = bih[0 * HDIM + gc] + bhh[0 * HDIM + gc];
  const float bi_f = bih[1 * HDIM + gc] + bhh[1 * HDIM + gc];
  const float bi_g = bih[2 * HDIM + gc] + bhh[2 * HDIM + gc];
  const float bi_o = bih[3 * HDIM + gc] + bhh[3 * HDIM + gc];
  float creg[4], hreg[4];
#pragma unroll
  for (int q = 0; q < 4; ++q) {
    creg[q] = c0[(b0 + 16 * q) * HDIM + gc];
    hreg[q] = h0[(b0 + 16 * q) * HDIM + gc];
  }
  const float a = alphap[0];
  __syncthreads();

  for (int t = 0; t < T_STEPS; ++t) {
    const unsigned short* xt    = xb + (size_t)t * (BATCH * IDIM);
    const unsigned short* hb    = hbuf + (size_t)(t & 1) * (BATCH * HDIM);
    unsigned short*       hnext = hbuf + (size_t)((t + 1) & 1) * (BATCH * HDIM);

    // pull next timestep's x tile toward the caches while we compute
    if (t + 1 < T_STEPS) {
      const unsigned short* xn = xb + (size_t)(t + 1) * (BATCH * IDIM);
      __builtin_prefetch(xn + (size_t)tid * 128, 0, 1);
    }

    v8f acc0 = {0.f, 0.f, 0.f, 0.f, 0.f, 0.f, 0.f, 0.f};
    v8f acc1 = {0.f, 0.f, 0.f, 0.f, 0.f, 0.f, 0.f, 0.f};

    // gates[64 x 64] = [x_t || h] @ Wslice^T,  K = 1024, 32 wmma k-steps
    for (int kit = 0; kit < KTOT / 32; ++kit) {
      const int kbase = kit * 32;
      const unsigned short* asrc;
      int kk;
      if (kbase < IDIM) { asrc = xt; kk = kbase; }
      else              { asrc = hb; kk = kbase - IDIM; }

      // A fragment (16x32 bf16): lanes<16 hold K{0..7,16..23}, lanes>=16 K{8..15,24..31}
      const unsigned short* ap = asrc + (size_t)(mtile * 16 + lr) * 512 + kk + hi * 8;
      v8bf alo = *reinterpret_cast<const v8bf*>(ap);
      v8bf ahi = *reinterpret_cast<const v8bf*>(ap + 16);
      v16bf afrag = __builtin_shufflevector(alo, ahi,
          0, 1, 2, 3, 4, 5, 6, 7, 8, 9, 10, 11, 12, 13, 14, 15);

      // B fragments (32x16 bf16) from LDS; padded stride -> conflict-free b128 reads
      const unsigned short* bp0 = &Wsl[(nb * 16 + lr) * WSTRIDE + kbase + hi * 8];
      const unsigned short* bp1 = &Wsl[((nb + 1) * 16 + lr) * WSTRIDE + kbase + hi * 8];
      v8bf b0lo = *reinterpret_cast<const v8bf*>(bp0);
      v8bf b0hi = *reinterpret_cast<const v8bf*>(bp0 + 16);
      v8bf b1lo = *reinterpret_cast<const v8bf*>(bp1);
      v8bf b1hi = *reinterpret_cast<const v8bf*>(bp1 + 16);
      v16bf bf0 = __builtin_shufflevector(b0lo, b0hi,
          0, 1, 2, 3, 4, 5, 6, 7, 8, 9, 10, 11, 12, 13, 14, 15);
      v16bf bf1 = __builtin_shufflevector(b1lo, b1hi,
          0, 1, 2, 3, 4, 5, 6, 7, 8, 9, 10, 11, 12, 13, 14, 15);

      acc0 = __builtin_amdgcn_wmma_f32_16x16x32_bf16(false, afrag, false, bf0,
                                                     (short)0, acc0, false, false);
      acc1 = __builtin_amdgcn_wmma_f32_16x16x32_bf16(false, afrag, false, bf1,
                                                     (short)0, acc1, false, false);
    }

    // spill gate tiles: D layout = VGPR r: (M=r, N=lane) / (M=8+r, N=lane-16)
    {
      const int row0 = mtile * 16 + hi * 8;
#pragma unroll
      for (int r = 0; r < 8; ++r) {
        gbuf[(row0 + r) * GST + nb * 16 + lr]       = acc0[r];
        gbuf[(row0 + r) * GST + (nb + 1) * 16 + lr] = acc1[r];
      }
    }
    __syncthreads();

    // fused elementwise: activations + alpha interpolation (state in registers)
#pragma unroll
    for (int q = 0; q < 4; ++q) {
      const int b = b0 + 16 * q;
      float gi = gbuf[b * GST + j]      + bi_i;
      float gf = gbuf[b * GST + 16 + j] + bi_f;
      float gg = gbuf[b * GST + 32 + j] + bi_g;
      float go = gbuf[b * GST + 48 + j] + bi_o;
      float si = fast_sigmoid(gi);
      float sf = fast_sigmoid(gf);
      float tg = fast_tanh(gg);
      float so = fast_sigmoid(go);
      float cn = sf * creg[q] + si * tg;
      float hn = so * fast_tanh(cn);
      float co = a * cn + (1.f - a) * creg[q];
      float ho = a * hn + (1.f - a) * hreg[q];
      creg[q] = co;
      hreg[q] = ho;
      hnext[b * HDIM + gc] = f2bf_rne(ho);
      // ys is 134MB write-once: non-temporal, don't evict L2-resident x/h
      __builtin_nontemporal_store(ho, &out[(size_t)t * (BATCH * HDIM) + b * HDIM + gc]);
    }
    __syncthreads();

    // device-wide step barrier: release h_next, wait for all 32 WGs
    if (tid == 0) {
      __hip_atomic_fetch_add(syncp, 1u, __ATOMIC_RELEASE, __HIP_MEMORY_SCOPE_AGENT);
      const unsigned int target = (unsigned int)NWG * (unsigned int)(t + 1);
      while (__hip_atomic_load(syncp, __ATOMIC_ACQUIRE, __HIP_MEMORY_SCOPE_AGENT) < target)
        __builtin_amdgcn_s_sleep(1);
    }
    __syncthreads();
  }

  // final hT / cT (post-interpolation carries, straight from registers)
  float* hT = out + (size_t)T_STEPS * BATCH * HDIM;
  float* cT = hT + BATCH * HDIM;
#pragma unroll
  for (int q = 0; q < 4; ++q) {
    const int b = b0 + 16 * q;
    hT[b * HDIM + gc] = hreg[q];
    cT[b * HDIM + gc] = creg[q];
  }
}

extern "C" void kernel_launch(void* const* d_in, const int* in_sizes, int n_in,
                              void* d_out, int out_size, void* d_ws, size_t ws_size,
                              hipStream_t stream) {
  const float* x     = (const float*)d_in[0];
  const float* h0    = (const float*)d_in[1];
  const float* c0    = (const float*)d_in[2];
  const float* Wih   = (const float*)d_in[3];
  const float* Whh   = (const float*)d_in[4];
  const float* bih   = (const float*)d_in[5];
  const float* bhh   = (const float*)d_in[6];
  const float* alpha = (const float*)d_in[7];
  float* out = (float*)d_out;

  // workspace layout: x_bf16 (64MB) | hbuf[2] bf16 (128KB) | sync counter
  const size_t xbytes = (size_t)T_STEPS * BATCH * IDIM * sizeof(unsigned short);
  unsigned short* xb   = (unsigned short*)d_ws;
  unsigned short* hbuf = (unsigned short*)((char*)d_ws + xbytes);
  unsigned int*   sync = (unsigned int*)((char*)d_ws + xbytes +
                                         2 * (size_t)BATCH * HDIM * sizeof(unsigned short));

  const int n4 = (T_STEPS * BATCH * IDIM) / 4;
  xcvt_kernel<<<(n4 + 255) / 256, 256, 0, stream>>>(x, xb, n4);
  hprep_kernel<<<((BATCH * HDIM / 4) + 255) / 256, 256, 0, stream>>>(h0, hbuf, sync);
  lstm_persistent<<<NWG, 256, 0, stream>>>(xb, h0, c0, Wih, Whh, bih, bhh, alpha,
                                           hbuf, sync, out);
}
